// MPNEncoder_58394375356589
// MI455X (gfx1250) — matrix operand order repaired and checked by reading, not compiled, not run on hardware
//
#include <hip/hip_runtime.h>

typedef __attribute__((ext_vector_type(16))) _Float16 v16h;
typedef __attribute__((ext_vector_type(8)))  float    v8f;

// Problem constants
constexpr int  EB = 32, EN = 64, EH = 300;
constexpr int  ABASE = 133, BBASE = 14, BFDIM = 193, OUTF = 465;
constexpr long ESTR = (long)EB * EN * EH;          // 614400 floats per per-encoding array

// Workspace float offsets (each 256B aligned)
constexpr long OFF_WA   = 0;                       // Wa  [300][133]
constexpr long OFF_WOA  = 39936;                   // Woa [300][133]
constexpr long OFF_WB   = 79872;                   // Wb  [300][14]
constexpr long OFF_WH16 = 84096;                   // Wh16 [320][320] f16 (51200 floats)
constexpr long OFF_A    = 135296;                  // then A,S0,R0,S1,T0,T1,AM each 2*ESTR

// ---------------------------------------------------------------- prep weights
__global__ __launch_bounds__(256) void prep_kernel(
    const float* __restrict__ W_i, const float* __restrict__ W_h,
    const float* __restrict__ W_o,
    float* __restrict__ Wa, float* __restrict__ Woa, float* __restrict__ Wb,
    _Float16* __restrict__ Wh16)
{
  int idx = blockIdx.x * blockDim.x + threadIdx.x;
  int stride = gridDim.x * blockDim.x;
  for (int i = idx; i < 300 * 133; i += stride) {
    int h = i / 133, f = i % 133;
    float wi = W_i[h * BFDIM + f];
    float wo = W_o[h * OUTF + f];
    if (f >= 101) { wi += W_i[h * BFDIM + 133 + (f - 101)];
                    wo += W_o[h * OUTF + 133 + (f - 101)]; }
    Wa[i] = wi; Woa[i] = wo;
  }
  for (int i = idx; i < 300 * 14; i += stride) {
    int h = i / 14, c = i % 14;
    Wb[i] = W_i[h * BFDIM + 165 + c] + W_i[h * BFDIM + 179 + c];
  }
  for (int i = idx; i < 320 * 320; i += stride) {
    int g = i / 320, h = i % 320;
    Wh16[i] = (g < 300 && h < 300) ? (_Float16)W_h[g * 300 + h] : (_Float16)0.f;
  }
}

// ---------------------------------------------------------------- A = fa @ Wa^T
__global__ __launch_bounds__(320) void atomA_kernel(
    const float* __restrict__ faR, const float* __restrict__ faP,
    const float* __restrict__ Wa, float* __restrict__ Abuf)
{
  int m = blockIdx.x, b = blockIdx.y, enc = blockIdx.z;
  const float* fa = (enc ? faP : faR) + ((long)b * EN + m) * ABASE;
  __shared__ float fas[ABASE];
  int tid = threadIdx.x;
  if (tid < ABASE) fas[tid] = fa[tid];
  __syncthreads();
  if (tid < EH) {
    float acc = 0.f;
    const float* wr = Wa + (long)tid * ABASE;
    for (int f = 0; f < ABASE; ++f) acc += fas[f] * wr[f];
    Abuf[(long)enc * ESTR + ((long)b * EN + m) * EH + tid] = acc;
  }
}

// ---------------------------------------------------------------- pass1: R0 (mode0) / S0 (mode1)
__global__ __launch_bounds__(320) void pass1_kernel(
    const float* __restrict__ fbR, const float* __restrict__ fbP,
    const float* __restrict__ adjR, const float* __restrict__ adjP,
    const float* __restrict__ Abuf, const float* __restrict__ Wb,
    float* __restrict__ Out, int mode)
{
  int i = blockIdx.x, b = blockIdx.y, enc = blockIdx.z;
  const float* fb  = enc ? fbP  : fbR;
  const float* adj = enc ? adjP : adjR;
  const float* A   = Abuf + (long)enc * ESTR;
  __shared__ float fbs[BBASE];
  int tid = threadIdx.x;
  float wbv[BBASE]; float acc = 0.f, arow = 0.f;
  if (tid < EH) {
    #pragma unroll
    for (int c = 0; c < BBASE; ++c) wbv[c] = Wb[tid * BBASE + c];
    if (!mode) arow = A[((long)b * EN + i) * EH + tid];
  }
  for (int j = 0; j < EN; ++j) {
    int m = mode ? j : i;
    int n = mode ? i : j;
    __syncthreads();
    if (tid < BBASE) fbs[tid] = fb[(((long)b * EN + m) * EN + n) * BBASE + tid];
    __syncthreads();
    if (tid < EH) {
      float a = mode ? A[((long)b * EN + j) * EH + tid] : arow;
      float d = a;
      #pragma unroll
      for (int c = 0; c < BBASE; ++c) d += fbs[c] * wbv[c];
      float v = adj[((long)b * EN + m) * EN + n] * d;
      acc += fmaxf(v, 0.f);
    }
  }
  if (tid < EH) Out[(long)enc * ESTR + ((long)b * EN + i) * EH + tid] = acc;
}

// ---------------------------------------------------------------- S1 = adj@S0 - R0
__global__ __launch_bounds__(320) void s1_kernel(
    const float* __restrict__ adjR, const float* __restrict__ adjP,
    const float* __restrict__ S0, const float* __restrict__ R0,
    float* __restrict__ S1)
{
  int n = blockIdx.x, b = blockIdx.y, enc = blockIdx.z;
  const float* adj = enc ? adjP : adjR;
  __shared__ float adjs[EN];
  int tid = threadIdx.x;
  if (tid < EN) adjs[tid] = adj[((long)b * EN + n) * EN + tid];
  __syncthreads();
  if (tid < EH) {
    float acc = -R0[(long)enc * ESTR + ((long)b * EN + n) * EH + tid];
    for (int k = 0; k < EN; ++k)
      acc += adjs[k] * S0[(long)enc * ESTR + ((long)b * EN + k) * EH + tid];
    S1[(long)enc * ESTR + ((long)b * EN + n) * EH + tid] = acc;
  }
}

// ---------------------------------------------------------------- T0 = S0@W_h^T, T1 = S1@W_h^T
__global__ __launch_bounds__(320) void t_kernel(
    const float* __restrict__ S0, const float* __restrict__ S1,
    const float* __restrict__ W_h,
    float* __restrict__ T0, float* __restrict__ T1)
{
  int n = blockIdx.x, b = blockIdx.y, enc = blockIdx.z;
  long row = (long)enc * ESTR + ((long)b * EN + n) * EH;
  __shared__ float s0s[EH], s1s[EH];
  int tid = threadIdx.x;
  if (tid < EH) { s0s[tid] = S0[row + tid]; s1s[tid] = S1[row + tid]; }
  __syncthreads();
  if (tid < EH) {
    float a0 = 0.f, a1 = 0.f;
    const float* wr = W_h + (long)tid * EH;
    for (int h = 0; h < EH; ++h) { float w = wr[h]; a0 += s0s[h] * w; a1 += s1s[h] * w; }
    T0[row + tid] = a0; T1[row + tid] = a1;
  }
}

// ---------------------------------------------------------------- pass2: the WMMA core
// block = (ntile, b, enc); 320 threads = 10 waves; wave w owns g in [32w, 32w+32)
__global__ __launch_bounds__(320) void pass2_kernel(
    const float* __restrict__ Abuf, const float* __restrict__ T0b,
    const float* __restrict__ T1b, float* __restrict__ AMb,
    const float* __restrict__ adjR, const float* __restrict__ adjP,
    const float* __restrict__ fbR, const float* __restrict__ fbP,
    const float* __restrict__ Wb, const _Float16* __restrict__ Wh16)
{
  int nt = blockIdx.x, b = blockIdx.y, enc = blockIdx.z;
  const float* adj = enc ? adjP : adjR;
  const float* fb  = enc ? fbP  : fbR;
  const float* A   = Abuf + (long)enc * ESTR;
  const float* T0  = T0b  + (long)enc * ESTR;
  const float* T1  = T1b  + (long)enc * ESTR;
  float* AM        = AMb  + (long)enc * ESTR;
  int n0 = nt * 16;

  __shared__ float    inp_s[16][304];
  __shared__ _Float16 m0_s[16][320];
  __shared__ float    arow_s[304];
  __shared__ float    fbrow_s[16][BBASE];
  __shared__ float    adjrow_s[16];   // adj[b][m][n0+r]
  __shared__ float    adjcol_s[16];   // adj[b][n0+r][m]

  int tid = threadIdx.x, w = tid >> 5, lane = tid & 31;
  int gcol = lane & 15, rh = lane >> 4;
  int g0 = w * 32;

  // Preload B = W_h^T f16 tiles into VGPRs (reused for all 64 m iterations).
  // B k order used here (contiguous, same as A): lane = (k<16?0:16)+n, v16h element e <-> K = 16*(lane>>4)+e
  v16h breg[2][10];
  #pragma unroll
  for (int t = 0; t < 2; ++t) {
    int g = g0 + 16 * t + gcol;
    #pragma unroll
    for (int kc = 0; kc < 10; ++kc) {
      int kb = rh * 16 + 32 * kc;
      const _Float16* p = Wh16 + (long)g * 320 + kb;
      v16h bv;
      #pragma unroll
      for (int e = 0; e < 16; ++e) bv[e] = p[e];
      breg[t][kc] = bv;
    }
  }

  // Preload T0 at this block's (n,g) footprint (C/D layout: lane=gcol+(r>=8)*16, vgpr=r%8)
  float t0r[2][8];
  #pragma unroll
  for (int t = 0; t < 2; ++t) {
    int g = g0 + 16 * t + gcol;
    #pragma unroll
    for (int v = 0; v < 8; ++v) {
      int r = rh * 8 + v;
      t0r[t][v] = (g < EH) ? T0[((long)b * EN + n0 + r) * EH + g] : 0.f;
    }
  }

  float wbv[BBASE];
  if (tid < EH) {
    #pragma unroll
    for (int c = 0; c < BBASE; ++c) wbv[c] = Wb[tid * BBASE + c];
  }

  v8f msum[2];
  #pragma unroll
  for (int t = 0; t < 2; ++t)
    #pragma unroll
    for (int e = 0; e < 8; ++e) msum[t][e] = 0.f;

  for (int m = 0; m < EN; ++m) {
    // ---- stage per-m rows
    if (tid < EH) arow_s[tid] = A[((long)b * EN + m) * EH + tid];
    if (tid < 16 * BBASE) {
      int r = tid / BBASE, c = tid % BBASE;
      fbrow_s[r][c] = fb[(((long)b * EN + m) * EN + n0 + r) * BBASE + c];
    } else if (tid < 16 * BBASE + 16) {
      int r = tid - 16 * BBASE;
      adjrow_s[r] = adj[((long)b * EN + m) * EN + n0 + r];
      adjcol_s[r] = adj[((long)b * EN + n0 + r) * EN + m];
    }
    __syncthreads();

    // ---- build inp (f32) and M0=relu(inp) (f16, K padded to 320) in LDS
    if (tid < EH) {
      int h = tid;
      #pragma unroll
      for (int r = 0; r < 16; ++r) {
        float d = arow_s[h];
        #pragma unroll
        for (int c = 0; c < BBASE; ++c) d += fbrow_s[r][c] * wbv[c];
        float val = adjrow_s[r] * d;
        inp_s[r][h] = val;
        m0_s[r][h] = (_Float16)fmaxf(val, 0.f);
      }
    } else {
      int h = tid;  // 300..319 pad
      #pragma unroll
      for (int r = 0; r < 16; ++r) m0_s[r][h] = (_Float16)0.f;
    }
    __syncthreads();

    // ---- Z tile = M0 @ W_h^T via WMMA (A layout: lane M=lane&15, two 8-f16 runs per lane)
    v8f acc[2];
    #pragma unroll
    for (int t = 0; t < 2; ++t)
      #pragma unroll
      for (int e = 0; e < 8; ++e) acc[t][e] = 0.f;

    int rr = lane & 15, kb8 = rh * 16;
    #pragma unroll
    for (int kc = 0; kc < 10; ++kc) {
      int h0 = kc * 32;
      const _Float16* p0 = &m0_s[rr][h0 + kb8];
      const _Float16* p1 = &m0_s[rr][h0 + 8 + kb8];
      v16h av;
      #pragma unroll
      for (int e = 0; e < 8; ++e) { av[e] = p0[e]; av[e + 8] = p1[e]; }
      acc[0] = __builtin_amdgcn_wmma_f32_16x16x32_f16(
          false, av, false, breg[0][kc], (short)0, acc[0], false, false);
      acc[1] = __builtin_amdgcn_wmma_f32_16x16x32_f16(
          false, av, false, breg[1][kc], (short)0, acc[1], false, false);
    }

    // ---- epilogue: relu(inp + Z + adj^T*T1 - adj*T0), accumulate over m
    float adjnm[8], adjmn[8];
    #pragma unroll
    for (int v = 0; v < 8; ++v) { int r = rh * 8 + v; adjnm[v] = adjcol_s[r]; adjmn[v] = adjrow_s[r]; }
    #pragma unroll
    for (int t = 0; t < 2; ++t) {
      int g = g0 + 16 * t + gcol;
      if (g < EH) {
        float t1g = T1[((long)b * EN + m) * EH + g];
        #pragma unroll
        for (int v = 0; v < 8; ++v) {
          int r = rh * 8 + v;
          float z = acc[t][v] + adjnm[v] * t1g - adjmn[v] * t0r[t][v];
          float pre = inp_s[r][g] + z;
          msum[t][v] += fmaxf(pre, 0.f);
        }
      }
    }
    __syncthreads();
  }

  #pragma unroll
  for (int t = 0; t < 2; ++t) {
    int g = g0 + 16 * t + gcol;
    if (g < EH) {
      #pragma unroll
      for (int v = 0; v < 8; ++v) {
        int r = rh * 8 + v;
        AM[((long)b * EN + n0 + r) * EH + g] = msum[t][v];
      }
    }
  }
}

// ---------------------------------------------------------------- output: relu([fa|a_msg]@W_o^T + b)
__global__ __launch_bounds__(320) void out_kernel(
    const float* __restrict__ faR, const float* __restrict__ faP,
    const float* __restrict__ AM, const float* __restrict__ Woa,
    const float* __restrict__ W_o, const float* __restrict__ W_ob,
    float* __restrict__ out)
{
  int n = blockIdx.x, b = blockIdx.y, enc = blockIdx.z;
  const float* fa = (enc ? faP : faR) + ((long)b * EN + n) * ABASE;
  const float* am = AM + (long)enc * ESTR + ((long)b * EN + n) * EH;
  __shared__ float fas[ABASE];
  __shared__ float ams[EH];
  int tid = threadIdx.x;
  if (tid < ABASE) fas[tid] = fa[tid];
  if (tid < EH)    ams[tid] = am[tid];
  __syncthreads();
  if (tid < EH) {
    int h = tid;
    float acc = W_ob[h];
    const float* wr1 = Woa + (long)h * ABASE;
    for (int f = 0; f < ABASE; ++f) acc += fas[f] * wr1[f];
    const float* wr2 = W_o + (long)h * OUTF + 165;
    for (int g = 0; g < EH; ++g) acc += ams[g] * wr2[g];
    out[(((long)enc * EB + b) * EN + n) * EH + h] = fmaxf(acc, 0.f);
  }
}

// ---------------------------------------------------------------- launch
extern "C" void kernel_launch(void* const* d_in, const int* in_sizes, int n_in,
                              void* d_out, int out_size, void* d_ws, size_t ws_size,
                              hipStream_t stream) {
  (void)in_sizes; (void)n_in; (void)out_size; (void)ws_size;
  const float* faR  = (const float*)d_in[0];
  const float* faP  = (const float*)d_in[1];
  const float* fbR  = (const float*)d_in[2];
  const float* fbP  = (const float*)d_in[3];
  const float* adjR = (const float*)d_in[4];
  const float* adjP = (const float*)d_in[5];
  const float* W_i  = (const float*)d_in[6];
  const float* W_h  = (const float*)d_in[7];
  const float* W_o  = (const float*)d_in[8];
  const float* W_ob = (const float*)d_in[9];
  float* out = (float*)d_out;

  float* ws = (float*)d_ws;
  float*    Wa   = ws + OFF_WA;
  float*    Woa  = ws + OFF_WOA;
  float*    Wbp  = ws + OFF_WB;
  _Float16* Wh16 = (_Float16*)(ws + OFF_WH16);
  float* Abuf = ws + OFF_A;
  float* S0 = Abuf + 2 * ESTR;
  float* R0 = S0 + 2 * ESTR;
  float* S1 = R0 + 2 * ESTR;
  float* T0 = S1 + 2 * ESTR;
  float* T1 = T0 + 2 * ESTR;
  float* AM = T1 + 2 * ESTR;

  prep_kernel<<<256, 256, 0, stream>>>(W_i, W_h, W_o, Wa, Woa, Wbp, Wh16);
  atomA_kernel<<<dim3(EN, EB, 2), 320, 0, stream>>>(faR, faP, Wa, Abuf);
  pass1_kernel<<<dim3(EN, EB, 2), 320, 0, stream>>>(fbR, fbP, adjR, adjP, Abuf, Wbp, R0, 0);
  pass1_kernel<<<dim3(EN, EB, 2), 320, 0, stream>>>(fbR, fbP, adjR, adjP, Abuf, Wbp, S0, 1);
  s1_kernel<<<dim3(EN, EB, 2), 320, 0, stream>>>(adjR, adjP, S0, R0, S1);
  t_kernel<<<dim3(EN, EB, 2), 320, 0, stream>>>(S0, S1, W_h, T0, T1);
  pass2_kernel<<<dim3(4, EB, 2), 320, 0, stream>>>(Abuf, T0, T1, AM, adjR, adjP, fbR, fbP, Wbp, Wh16);
  out_kernel<<<dim3(EN, EB, 2), 320, 0, stream>>>(faR, faP, AM, Woa, W_o, W_ob, out);
}
